// Attention_58703613002311
// MI455X (gfx1250) — matrix operand order repaired and checked by reading, not compiled, not run on hardware
//
#include <hip/hip_runtime.h>
#include <hip/hip_bf16.h>

#define DIM    2048
#define SEQ    4096
#define NH     16
#define NKV    4
#define DQK    128
#define DV     128
#define WIN    512
#define FRONT  256
#define NQKV   3072   // 2048 Q cols + 512 K cols + 512 V cols
#define HALF_D 64

typedef __attribute__((ext_vector_type(16))) __bf16 bf16x16;
typedef __attribute__((ext_vector_type(8)))  __bf16 bf16x8;
typedef __attribute__((ext_vector_type(8)))  float  f32x8;
typedef __attribute__((ext_vector_type(4)))  unsigned int u32x4;

union Frag  { bf16x16 v; u32x4 q[2]; };
union Half8 { bf16x8  v; u32x4 q;    };

__device__ __forceinline__ u32x4 ld128(const __bf16* p) {
  return *reinterpret_cast<const u32x4*>(p);
}

__device__ __forceinline__ f32x8 wmma_bf16(const bf16x16& a, const bf16x16& b, const f32x8& c) {
  return __builtin_amdgcn_wmma_f32_16x16x32_bf16(false, a, false, b, (short)0, c, false, false);
}

// ---------------------------------------------------------------- RoPE tables
__global__ void k_rope_tables(float* __restrict__ ct, float* __restrict__ st) {
  int idx = blockIdx.x * blockDim.x + threadIdx.x;
  if (idx >= SEQ * HALF_D) return;
  int t = idx >> 6;
  int f = idx & (HALF_D - 1);
  float inv_freq = __powf(10000.0f, -((float)(2 * f) / (float)DQK));
  float ang = (float)t * inv_freq;
  ct[idx] = cosf(ang);
  st[idx] = sinf(ang);
}

// ---------------------------------------------------------------- f32 -> bf16
__global__ void k_cvt_x(const float* __restrict__ x, __bf16* __restrict__ xb, int n) {
  int i = blockIdx.x * blockDim.x + threadIdx.x;
  if (i < n) xb[i] = (__bf16)x[i];
}

// Build [wq|wk|wv]^T as bf16, shape (NQKV, DIM): row n = output feature, col k.
__global__ void k_build_wqkvT(const float* __restrict__ wq, const float* __restrict__ wk,
                              const float* __restrict__ wv, __bf16* __restrict__ wt) {
  int i = blockIdx.x * blockDim.x + threadIdx.x;
  if (i >= NQKV * DIM) return;
  int n = i / DIM, k = i - n * DIM;
  float v;
  if (n < DIM)            v = wq[(size_t)k * DIM + n];
  else if (n < DIM + 512) v = wk[(size_t)k * 512 + (n - DIM)];
  else                    v = wv[(size_t)k * 512 + (n - DIM - 512)];
  wt[i] = (__bf16)v;
}

// wo^T as bf16, shape (DIM, DIM)
__global__ void k_build_woT(const float* __restrict__ wo, __bf16* __restrict__ wt) {
  int i = blockIdx.x * blockDim.x + threadIdx.x;
  if (i >= DIM * DIM) return;
  int n = i / DIM, k = i - n * DIM;
  wt[i] = (__bf16)wo[(size_t)k * DIM + n];
}

// ------------------------------------------------- fused QKV GEMM + RoPE + layout
// Wave tile: 32 rows x 128 cols. Grid: (SEQ/32) * (NQKV/128) waves.
__global__ void __launch_bounds__(128, 1)
k_qkv(const __bf16* __restrict__ xb, const __bf16* __restrict__ wt,
      const float* __restrict__ ct, const float* __restrict__ st,
      __bf16* __restrict__ Qb, __bf16* __restrict__ Kb,
      __bf16* __restrict__ Vt) {
  const int gid  = (blockIdx.x * blockDim.x + threadIdx.x) >> 5;
  const int lane = threadIdx.x & 31;
  const int lo = lane & 15, hi = lane >> 4;
  const int mi = gid % (SEQ / 32);
  const int ni = gid / (SEQ / 32);
  const int mbase = mi * 32;
  const int nbase = ni * 128;

  const f32x8 zero = {};
  f32x8 acc[2][8];
#pragma unroll
  for (int ms = 0; ms < 2; ++ms)
#pragma unroll
    for (int nt = 0; nt < 8; ++nt) acc[ms][nt] = zero;

  // single base pointers; all subtile addressing via constant immediate offsets
  const __bf16* pa = xb + (size_t)(mbase + lo) * DIM + hi * 8;
  const __bf16* pb = wt + (size_t)(nbase + lo) * DIM + hi * 8;

#pragma unroll 2
  for (int kc = 0; kc < DIM; kc += 32) {
    Frag a0, a1;
    a0.q[0] = ld128(pa);
    a0.q[1] = ld128(pa + 16);
    a1.q[0] = ld128(pa + 16 * DIM);
    a1.q[1] = ld128(pa + 16 * DIM + 16);
#pragma unroll
    for (int nt = 0; nt < 8; ++nt) {
      Frag b;
      b.q[0] = ld128(pb + nt * 16 * DIM);
      b.q[1] = ld128(pb + nt * 16 * DIM + 16);
      acc[0][nt] = wmma_bf16(a0.v, b.v, acc[0][nt]);
      acc[1][nt] = wmma_bf16(a1.v, b.v, acc[1][nt]);
    }
    pa += 32;
    pb += 32;
  }

  if (nbase < DIM + NKV * DQK) {
    // ---- Q or K region: apply RoPE then store row-major bf16
    const bool isQ = (nbase < DIM);
    const int  cb  = isQ ? nbase : (nbase - DIM);  // column base within Q or K buffer
    const int  ldc = isQ ? DIM : (NKV * DQK);
    __bf16* dst = isQ ? Qb : Kb;
#pragma unroll
    for (int ms = 0; ms < 2; ++ms) {
#pragma unroll
      for (int nt = 0; nt < 8; ++nt) {
        const int d = nt * 16 + lo;      // dim within this head (tile == one head)
        const int f = d >> 1;
#pragma unroll
        for (int r = 0; r < 8; ++r) {
          const int row = mbase + ms * 16 + r + 8 * hi;
          float v  = acc[ms][nt][r];
          float pv = __shfl_xor(v, 1, 32);     // RoPE pair partner (adjacent lane)
          float c  = ct[row * HALF_D + f];
          float s  = st[row * HALF_D + f];
          float o  = (d & 1) ? (pv * s + v * c) : (v * c - pv * s);
          dst[(size_t)row * ldc + cb + d] = (__bf16)o;
        }
      }
    }
  } else {
    // ---- V region: store transposed Vt[kvh][d][seq]
    const int cb  = nbase - (DIM + NKV * DQK);   // multiple of 128
    const int kvh = cb / DV;
#pragma unroll
    for (int ms = 0; ms < 2; ++ms) {
#pragma unroll
      for (int nt = 0; nt < 8; ++nt) {
        const int d = nt * 16 + lo;
        Half8 h;
#pragma unroll
        for (int r = 0; r < 8; ++r) h.v[r] = (__bf16)acc[ms][nt][r];
        const size_t off = (size_t)kvh * DV * SEQ + (size_t)d * SEQ
                         + mbase + ms * 16 + 8 * hi;   // 8 consecutive seq positions
        *reinterpret_cast<u32x4*>(Vt + off) = h.q;
      }
    }
  }
}

// ------------------------------------------------------- flash attention
// One wave per (head, 16-query tile). Grid: NH * SEQ/16 waves, 4 waves/block.
__global__ void __launch_bounds__(128, 1)
k_attn(const __bf16* __restrict__ Qb, const __bf16* __restrict__ Kb,
       const __bf16* __restrict__ Vt, __bf16* __restrict__ AO) {
  __shared__ __bf16 pbuf[4][16 * 32];
  const int wib  = threadIdx.x >> 5;
  const int gid  = (blockIdx.x * blockDim.x + threadIdx.x) >> 5;
  const int lane = threadIdx.x & 31;
  const int lo = lane & 15, hi = lane >> 4;
  const int head = gid / (SEQ / 16);
  const int qb16 = (gid % (SEQ / 16)) * 16;
  const int kvh  = head / (NH / NKV);
  const float scale = 0.08838834764831845f;   // 1/sqrt(128)
  const float NEGB  = -1e30f;

  // resident Q fragments (A layout), 4 K-steps of 32 covering d=128
  Frag qf[4];
  {
    const __bf16* pq = Qb + (size_t)(qb16 + lo) * DIM + head * DQK + hi * 8;
#pragma unroll
    for (int c = 0; c < 4; ++c) {
      qf[c].q[0] = ld128(pq + c * 32);
      qf[c].q[1] = ld128(pq + c * 32 + 16);
    }
  }

  const f32x8 zero = {};
  f32x8 O[8];
  float m[8], l[8];
#pragma unroll
  for (int nt = 0; nt < 8; ++nt) O[nt] = zero;
#pragma unroll
  for (int r = 0; r < 8; ++r) { m[r] = NEGB; l[r] = 0.0f; }

  // invariant pieces of the K / V base addresses
  const __bf16* pk0 = Kb + (size_t)lo * (NKV * DQK) + kvh * DQK + hi * 8;
  const __bf16* pv0 = Vt + (size_t)kvh * DV * SEQ + (size_t)lo * SEQ + hi * 8;

  const int qmax = qb16 + 15;
  for (int kb = 0; kb < qb16 + 16; kb += 32) {
    const bool band  = (kb + 31 >= qb16 - (WIN - 1));
    const bool front = (kb < FRONT) && (kb <= qmax - WIN);
    if (!(band || front)) continue;

    // ---- scores: two 16x16 tiles over d=128
    const __bf16* pk = pk0 + (size_t)kb * (NKV * DQK);
    f32x8 S0 = zero, S1 = zero;
#pragma unroll
    for (int c = 0; c < 4; ++c) {
      Frag b;
      b.q[0] = ld128(pk + c * 32);
      b.q[1] = ld128(pk + c * 32 + 16);
      S0 = wmma_bf16(qf[c].v, b.v, S0);
      b.q[0] = ld128(pk + 16 * (NKV * DQK) + c * 32);
      b.q[1] = ld128(pk + 16 * (NKV * DQK) + c * 32 + 16);
      S1 = wmma_bf16(qf[c].v, b.v, S1);
    }

    // ---- mask + online softmax update (per-row across the two subtiles)
    const int j0 = kb + lo, j1 = kb + 16 + lo;
#pragma unroll
    for (int r = 0; r < 8; ++r) {
      const int i = qb16 + r + 8 * hi;
      float s0 = S0[r] * scale, s1 = S1[r] * scale;
      const bool a0 = (j0 <= i && j0 > i - WIN) || (j0 < FRONT && j0 <= i - WIN);
      const bool a1 = (j1 <= i && j1 > i - WIN) || (j1 < FRONT && j1 <= i - WIN);
      s0 = a0 ? s0 : NEGB;
      s1 = a1 ? s1 : NEGB;
      float bm = fmaxf(s0, s1);
#pragma unroll
      for (int off = 1; off < 16; off <<= 1) bm = fmaxf(bm, __shfl_xor(bm, off, 32));
      const float mnew = fmaxf(m[r], bm);
      const float corr = __expf(m[r] - mnew);
      const float p0 = __expf(s0 - mnew);
      const float p1 = __expf(s1 - mnew);
      float rs = p0 + p1;
#pragma unroll
      for (int off = 1; off < 16; off <<= 1) rs += __shfl_xor(rs, off, 32);
      l[r] = l[r] * corr + rs;
      m[r] = mnew;
#pragma unroll
      for (int nt = 0; nt < 8; ++nt) O[nt][r] *= corr;
      const int prow = r + 8 * hi;
      pbuf[wib][prow * 32 + lo]      = (__bf16)p0;
      pbuf[wib][prow * 32 + 16 + lo] = (__bf16)p1;
    }

    // ---- reload P in A layout from LDS (same-wave DS ordering)
    Frag pf;
    {
      const __bf16* p = &pbuf[wib][lo * 32 + hi * 8];
      pf.q[0] = *reinterpret_cast<const u32x4*>(p);
      pf.q[1] = *reinterpret_cast<const u32x4*>(p + 16);
    }

    // ---- O += P @ V  (V transposed: contiguous keys per dv column)
    const __bf16* pv = pv0 + kb;
#pragma unroll
    for (int nt = 0; nt < 8; ++nt) {
      Frag b;
      b.q[0] = ld128(pv + nt * 16 * SEQ);
      b.q[1] = ld128(pv + nt * 16 * SEQ + 16);
      O[nt] = wmma_bf16(pf.v, b.v, O[nt]);
    }
  }

  // ---- normalize and store bf16 for the output projection
  float inv[8];
#pragma unroll
  for (int r = 0; r < 8; ++r) inv[r] = 1.0f / l[r];
#pragma unroll
  for (int nt = 0; nt < 8; ++nt) {
#pragma unroll
    for (int r = 0; r < 8; ++r) {
      const int row = qb16 + r + 8 * hi;
      AO[(size_t)row * DIM + head * DQK + nt * 16 + lo] = (__bf16)(O[nt][r] * inv[r]);
    }
  }
}

// ------------------------------------------------------- output projection GEMM
__global__ void __launch_bounds__(128, 1)
k_out(const __bf16* __restrict__ AO, const __bf16* __restrict__ wt,
      float* __restrict__ out) {
  const int gid  = (blockIdx.x * blockDim.x + threadIdx.x) >> 5;
  const int lane = threadIdx.x & 31;
  const int lo = lane & 15, hi = lane >> 4;
  const int mi = gid % (SEQ / 32);
  const int ni = gid / (SEQ / 32);
  const int mbase = mi * 32;
  const int nbase = ni * 128;

  const f32x8 zero = {};
  f32x8 acc[2][8];
#pragma unroll
  for (int ms = 0; ms < 2; ++ms)
#pragma unroll
    for (int nt = 0; nt < 8; ++nt) acc[ms][nt] = zero;

  const __bf16* pa = AO + (size_t)(mbase + lo) * DIM + hi * 8;
  const __bf16* pb = wt + (size_t)(nbase + lo) * DIM + hi * 8;

#pragma unroll 2
  for (int kc = 0; kc < DIM; kc += 32) {
    Frag a0, a1;
    a0.q[0] = ld128(pa);
    a0.q[1] = ld128(pa + 16);
    a1.q[0] = ld128(pa + 16 * DIM);
    a1.q[1] = ld128(pa + 16 * DIM + 16);
#pragma unroll
    for (int nt = 0; nt < 8; ++nt) {
      Frag b;
      b.q[0] = ld128(pb + nt * 16 * DIM);
      b.q[1] = ld128(pb + nt * 16 * DIM + 16);
      acc[0][nt] = wmma_bf16(a0.v, b.v, acc[0][nt]);
      acc[1][nt] = wmma_bf16(a1.v, b.v, acc[1][nt]);
    }
    pa += 32;
    pb += 32;
  }

#pragma unroll
  for (int ms = 0; ms < 2; ++ms)
#pragma unroll
    for (int nt = 0; nt < 8; ++nt)
#pragma unroll
      for (int r = 0; r < 8; ++r) {
        const int row = mbase + ms * 16 + r + 8 * hi;
        out[(size_t)row * DIM + nbase + nt * 16 + lo] = acc[ms][nt][r];
      }
}

// ---------------------------------------------------------------- launch
extern "C" void kernel_launch(void* const* d_in, const int* in_sizes, int n_in,
                              void* d_out, int out_size, void* d_ws, size_t ws_size,
                              hipStream_t stream) {
  const float* x  = (const float*)d_in[0];
  const float* wq = (const float*)d_in[1];
  const float* wk = (const float*)d_in[2];
  const float* wv = (const float*)d_in[3];
  const float* wo = (const float*)d_in[4];
  float* out = (float*)d_out;

  char* ws = (char*)d_ws;
  size_t off = 0;
  auto alloc = [&](size_t bytes) {
    char* p = ws + off;
    off += (bytes + 255) & ~(size_t)255;
    return p;
  };
  __bf16* xb    = (__bf16*)alloc((size_t)SEQ * DIM * 2);
  __bf16* wqkvT = (__bf16*)alloc((size_t)NQKV * DIM * 2);
  __bf16* woT   = (__bf16*)alloc((size_t)DIM * DIM * 2);
  __bf16* Qb    = (__bf16*)alloc((size_t)SEQ * DIM * 2);
  __bf16* Kb    = (__bf16*)alloc((size_t)SEQ * NKV * DQK * 2);
  __bf16* Vt    = (__bf16*)alloc((size_t)NKV * DV * SEQ * 2);
  float*  ct    = (float*)alloc((size_t)SEQ * HALF_D * 4);
  float*  st    = (float*)alloc((size_t)SEQ * HALF_D * 4);
  __bf16* AO    = (__bf16*)alloc((size_t)SEQ * DIM * 2);

  k_rope_tables<<<(SEQ * HALF_D + 255) / 256, 256, 0, stream>>>(ct, st);
  k_cvt_x<<<(SEQ * DIM + 255) / 256, 256, 0, stream>>>(x, xb, SEQ * DIM);
  k_build_wqkvT<<<(NQKV * DIM + 255) / 256, 256, 0, stream>>>(wq, wk, wv, wqkvT);
  k_build_woT<<<(DIM * DIM + 255) / 256, 256, 0, stream>>>(wo, woT);

  // QKV GEMM: (SEQ/32)*(NQKV/128) = 128*24 = 3072 waves, 4 waves/block
  k_qkv<<<(128 * 24) / 4, 128, 0, stream>>>(xb, wqkvT, ct, st, Qb, Kb, Vt);
  // Attention: NH*(SEQ/16) = 4096 waves
  k_attn<<<(NH * (SEQ / 16)) / 4, 128, 0, stream>>>(Qb, Kb, Vt, AO);
  // Output GEMM: (SEQ/32)*(DIM/128) = 128*16 = 2048 waves
  k_out<<<(128 * 16) / 4, 128, 0, stream>>>(AO, woT, out);
}
